// DocumentRelationAttention_74526272520876
// MI455X (gfx1250) — matrix operand order repaired and checked by reading, not compiled, not run on hardware
//
#include <hip/hip_runtime.h>
#include <math.h>

// ---------------------------------------------------------------------------
// CDNA5 (gfx1250) wave32 WMMA types
// ---------------------------------------------------------------------------
typedef __attribute__((ext_vector_type(16))) __bf16 v16bf;
typedef __attribute__((ext_vector_type(8)))  __bf16 v8bf;
typedef __attribute__((ext_vector_type(8)))  float  v8f;

union Frag16 { v16bf v; v8bf h[2]; };

__device__ __forceinline__ unsigned short f2bf(float f) {
  unsigned u = __float_as_uint(f);
  u += 0x7fffu + ((u >> 16) & 1u);          // round-to-nearest-even
  return (unsigned short)(u >> 16);
}
__device__ __forceinline__ float bf2f(unsigned short h) {
  return __uint_as_float(((unsigned)h) << 16);
}

__device__ __forceinline__ v8f wmma_bf16(const Frag16& a, const Frag16& b, v8f c) {
  return __builtin_amdgcn_wmma_f32_16x16x32_bf16(false, a.v, false, b.v,
                                                 (short)0, c, false, false);
}

// ---- CDNA5 async global->LDS path (guarded; fallback = reg-staged pipeline)
#if defined(__has_builtin)
#if __has_builtin(__builtin_amdgcn_global_load_async_to_lds_b128)
#define HAVE_ASYNC_LDS 1
#endif
#endif
#ifndef HAVE_ASYNC_LDS
#define HAVE_ASYNC_LDS 0
#endif

#if HAVE_ASYNC_LDS
typedef int v4i32 __attribute__((vector_size(16)));
typedef __attribute__((address_space(1))) v4i32* as1_v4p;
typedef __attribute__((address_space(3))) v4i32* as3_v4p;
__device__ __forceinline__ void async_copy16(const void* g, void* l) {
  // lane-wise 16B DMA into LDS; flat-shared low 32 bits == LDS offset
  __builtin_amdgcn_global_load_async_to_lds_b128(
      (as1_v4p)(unsigned long long)g,
      (as3_v4p)(unsigned)(unsigned long long)l, 0, 0);
}
#endif
__device__ __forceinline__ void wait_async0() {
#if defined(__has_builtin)
#if __has_builtin(__builtin_amdgcn_s_wait_asynccnt)
  __builtin_amdgcn_s_wait_asynccnt(0);
  return;
#endif
#endif
  asm volatile("s_wait_asynccnt 0" ::: "memory");
}

#define HNUM 12
#define DMODEL 768
#define DHEAD 64

// ---------------------------------------------------------------------------
// elementwise f32 -> bf16
// ---------------------------------------------------------------------------
__global__ void cvt_bf16_kernel(const float* __restrict__ src,
                                unsigned short* __restrict__ dst, int n) {
  int i = blockIdx.x * 256 + threadIdx.x;
  if (i < n) dst[i] = f2bf(src[i]);
}

// W [768,768] row-major (K x N) -> WT bf16 [768,768] (N x K)
__global__ void cvt_T_bf16_kernel(const float* __restrict__ W,
                                  unsigned short* __restrict__ WT) {
  int i = blockIdx.x * 256 + threadIdx.x;
  if (i < DMODEL * DMODEL) {
    int k = i / DMODEL, n = i % DMODEL;
    WT[n * DMODEL + k] = f2bf(W[i]);
  }
}

// ---------------------------------------------------------------------------
// WMMA GEMM: C[m,n] = sum_k A[m,k] * BT[n,k]   (A, BT bf16; K = N = 768)
// block 256 thr = 8 waves, tile 128x128, K-step 32, double-buffered LDS.
// mode 1: store bf16 head-major [b,h,l,d] (Lrows tokens per batch)
// mode 2: store f32 row-major [M,768]
// ---------------------------------------------------------------------------
__launch_bounds__(256)
__global__ void gemm_bf16_kernel(const unsigned short* __restrict__ A,
                                 const unsigned short* __restrict__ BT,
                                 void* __restrict__ C,
                                 int M, int mode, int Lrows) {
  const int K = DMODEL;
  __shared__ unsigned short Asm[2][128 * 40];
  __shared__ unsigned short Bsm[2][128 * 40];

  const int tid  = threadIdx.x;
  const int wave = tid >> 5, lane = tid & 31;
  const int l16 = lane & 15, hi = lane >> 4;
  const int wm = (wave >> 2) * 64;   // 0 / 64
  const int wn = (wave & 3) * 32;    // 0..96
  const int bm = blockIdx.x * 128, bn = blockIdx.y * 128;

  // per-thread staging coordinates: 2 b128 quads per 128x32 tile
  const int r0 = tid >> 2, c0 = (tid & 3) * 8;       // quad0: row r0
  const int gr0 = bm + r0, gr1 = bm + r0 + 64;       // quad1: row r0+64
  const unsigned short* Ap0 = A  + (size_t)gr0 * K + c0;
  const unsigned short* Ap1 = A  + (size_t)gr1 * K + c0;
  const unsigned short* Bp0 = BT + (size_t)(bn + r0) * K + c0;
  const unsigned short* Bp1 = BT + (size_t)(bn + r0 + 64) * K + c0;
  const bool a0ok = gr0 < M, a1ok = gr1 < M;
  const int la0 = r0 * 40 + c0, la1 = (r0 + 64) * 40 + c0;

  v8f acc[4][2];
  const v8f zacc = {0.f,0.f,0.f,0.f,0.f,0.f,0.f,0.f};
  for (int mi = 0; mi < 4; ++mi)
    for (int ni = 0; ni < 2; ++ni) acc[mi][ni] = zacc;

#if HAVE_ASYNC_LDS
  auto stage = [&](int k0, int buf) {
    const uint4 z = {0u, 0u, 0u, 0u};
    if (a0ok) async_copy16(Ap0 + k0, &Asm[buf][la0]);
    else      *(uint4*)&Asm[buf][la0] = z;
    if (a1ok) async_copy16(Ap1 + k0, &Asm[buf][la1]);
    else      *(uint4*)&Asm[buf][la1] = z;
    async_copy16(Bp0 + k0, &Bsm[buf][la0]);
    async_copy16(Bp1 + k0, &Bsm[buf][la1]);
  };
  stage(0, 0);
  wait_async0();
  __syncthreads();
#else
  uint4 ra0, ra1, rb0, rb1;
  auto gload = [&](int k0) {
    const uint4 z = {0u, 0u, 0u, 0u};
    ra0 = a0ok ? *(const uint4*)(Ap0 + k0) : z;
    ra1 = a1ok ? *(const uint4*)(Ap1 + k0) : z;
    rb0 = *(const uint4*)(Bp0 + k0);
    rb1 = *(const uint4*)(Bp1 + k0);
    if (k0 + 32 < K) {                       // near-scope speculative prefetch
      if (a0ok) __builtin_prefetch(Ap0 + k0 + 32, 0, 3);
      __builtin_prefetch(Bp0 + k0 + 32, 0, 3);
    }
  };
  auto lstore = [&](int buf) {
    *(uint4*)&Asm[buf][la0] = ra0;
    *(uint4*)&Asm[buf][la1] = ra1;
    *(uint4*)&Bsm[buf][la0] = rb0;
    *(uint4*)&Bsm[buf][la1] = rb1;
  };
  gload(0);
  lstore(0);
  __syncthreads();
#endif

  int buf = 0;
  for (int k0 = 0; k0 < K; k0 += 32, buf ^= 1) {
    const bool more = (k0 + 32) < K;
#if HAVE_ASYNC_LDS
    if (more) stage(k0 + 32, buf ^ 1);       // DMA next stage during math
#else
    if (more) gload(k0 + 32);                // issue next-stage global loads
#endif

    Frag16 fa[4], fb[2];
    for (int mi = 0; mi < 4; ++mi) {
      int row = wm + mi * 16 + l16;
      int kb = hi * 8;
      fa[mi].h[0] = *(const v8bf*)&Asm[buf][row * 40 + kb];
      fa[mi].h[1] = *(const v8bf*)&Asm[buf][row * 40 + kb + 16];
    }
    for (int ni = 0; ni < 2; ++ni) {
      int nrow = wn + ni * 16 + l16;
      int kb2 = hi * 8;
      fb[ni].h[0] = *(const v8bf*)&Bsm[buf][nrow * 40 + kb2];
      fb[ni].h[1] = *(const v8bf*)&Bsm[buf][nrow * 40 + kb2 + 16];
    }
    for (int mi = 0; mi < 4; ++mi)
      for (int ni = 0; ni < 2; ++ni)
        acc[mi][ni] = wmma_bf16(fa[mi], fb[ni], acc[mi][ni]);

#if HAVE_ASYNC_LDS
    if (more) wait_async0();                 // my lanes' DMAs landed
#else
    if (more) lstore(buf ^ 1);               // drain regs into next stage
#endif
    __syncthreads();                         // stage swap
  }

  // epilogue: C layout -> row m = base + vgpr + 8*hi, col n = base + l16
  for (int mi = 0; mi < 4; ++mi)
    for (int ni = 0; ni < 2; ++ni) {
      int n = bn + wn + ni * 16 + l16;
      for (int i = 0; i < 8; ++i) {
        int m = bm + wm + mi * 16 + hi * 8 + i;
        if (m >= M) continue;
        float v = acc[mi][ni][i];
        if (mode == 1) {
          int bI = m / Lrows, l = m % Lrows;
          int h = n >> 6, d = n & 63;
          ((unsigned short*)C)[(((size_t)(bI * HNUM + h) * Lrows + l) * DHEAD) + d] = f2bf(v);
        } else {
          ((float*)C)[(size_t)m * DMODEL + n] = v;
        }
      }
    }
}

// ---------------------------------------------------------------------------
// per-(b,h) sums over relation rows:  rksum/rvsum [48,64]
// ---------------------------------------------------------------------------
__global__ void rel_sums_kernel(const unsigned short* __restrict__ RK,
                                const unsigned short* __restrict__ RV,
                                float* __restrict__ rksum,
                                float* __restrict__ rvsum, int R) {
  int bh = blockIdx.x, d = threadIdx.x;   // 64 threads
  float sk = 0.f, sv = 0.f;
  for (int r = 0; r < R; ++r) {
    size_t idx = ((size_t)bh * R + r) * DHEAD + d;
    sk += bf2f(RK[idx]);
    sv += bf2f(RV[idx]);
  }
  rksum[bh * DHEAD + d] = sk;
  rvsum[bh * DHEAD + d] = sv;
}

// fusedK = K + rksum, fusedV = V + rvsum  (head-major bf16, 48*1024*64 elems)
__global__ void fuse_kv_kernel(const unsigned short* __restrict__ Kh,
                               const unsigned short* __restrict__ Vh,
                               const float* __restrict__ rksum,
                               const float* __restrict__ rvsum,
                               unsigned short* __restrict__ FK,
                               unsigned short* __restrict__ FV, int total) {
  int i = blockIdx.x * 256 + threadIdx.x;
  if (i < total) {
    int d = i & 63;
    int bh = i >> 16;                       // 1024*64 = 65536 per (b,h)
    FK[i] = f2bf(bf2f(Kh[i]) + rksum[bh * DHEAD + d]);
    FV[i] = f2bf(bf2f(Vh[i]) + rvsum[bh * DHEAD + d]);
  }
}

// ---------------------------------------------------------------------------
// Flash-style attention (wave32 WMMA):
//   Q head-major [BH, Mq, 64], K/V head-major [BH, 1024, 64] (bf16)
//   out = softmax(Q K^T * scale + SMOOTH, mask) @ V  (+optional Vres)
//   written token-major bf16 [b*Mq, 768] at column h*64+d
// block = 128 threads (4 waves), 64 query rows, 64-key tiles.
// ---------------------------------------------------------------------------
__launch_bounds__(128)
__global__ void flash_attn_kernel(const unsigned short* __restrict__ Q,
                                  const unsigned short* __restrict__ Kh,
                                  const unsigned short* __restrict__ Vh,
                                  const int* __restrict__ mask,
                                  const unsigned short* __restrict__ Vres,
                                  unsigned short* __restrict__ Out,
                                  int Mq) {
  const int Lk = 1024;
  __shared__ unsigned short Qs[64 * 72];
  __shared__ unsigned short Ks[64 * 72];
  __shared__ unsigned short VTs[64 * 72];
  __shared__ unsigned short Ps[64 * 72];
  __shared__ float Ss[64 * 66];
  __shared__ float factor_s[64];
  __shared__ float rsum_s[64];
  __shared__ int   mask_s[64];

  const int tid = threadIdx.x, wave = tid >> 5, lane = tid & 31;
  const int l16 = lane & 15, hi = lane >> 4;
  const int bh = blockIdx.y, b = bh / HNUM, h = bh % HNUM;
  const int m0 = blockIdx.x * 64;
  const int mo = wave * 16;

  const unsigned short* Qb = Q  + (size_t)bh * Mq * DHEAD;
  const unsigned short* Kb = Kh + (size_t)bh * Lk * DHEAD;
  const unsigned short* Vb = Vh + (size_t)bh * Lk * DHEAD;

  // stage Q tile (64x64 bf16), zero-padded past Mq
  for (int q = tid; q < 512; q += 128) {
    int r = q >> 3, c = (q & 7) * 8;
    int gm = m0 + r;
    uint4 va = {0u, 0u, 0u, 0u};
    if (gm < Mq) va = *(const uint4*)&Qb[(size_t)gm * DHEAD + c];
    *(uint4*)&Qs[r * 72 + c] = va;
  }

  const v8f zacc = {0.f,0.f,0.f,0.f,0.f,0.f,0.f,0.f};
  v8f accO[4];
  for (int dt = 0; dt < 4; ++dt) accO[dt] = zacc;
  float m_run = -3.0e38f, s_run = 0.0f;    // meaningful for tid < 64

  for (int j0 = 0; j0 < Lk; j0 += 64) {
    __syncthreads();
    // K tile (64 keys x 64 d), row-major == B^T layout for S = Q K^T
    for (int q = tid; q < 512; q += 128) {
      int r = q >> 3, c = (q & 7) * 8;
      *(uint4*)&Ks[r * 72 + c] = *(const uint4*)&Kb[(size_t)(j0 + r) * DHEAD + c];
    }
    // V tile transposed: VTs[d][j] = V[j0+j][d]  (B^T layout for O += P V)
    for (int e = tid; e < 4096; e += 128) {
      int j = e >> 6, d = e & 63;
      VTs[d * 72 + j] = Vb[(size_t)(j0 + j) * DHEAD + d];
    }
    if (tid < 64) mask_s[tid] = mask[b * Lk + j0 + tid];
    __syncthreads();

    // ---- S = Q K^T (each wave owns rows mo..mo+15, full 64 keys) ----
    v8f accS[4];
    for (int jt = 0; jt < 4; ++jt) accS[jt] = zacc;
    Frag16 fa[2];
    for (int ks = 0; ks < 2; ++ks) {
      int row = mo + l16, kb = hi * 8 + ks * 32;
      fa[ks].h[0] = *(const v8bf*)&Qs[row * 72 + kb];
      fa[ks].h[1] = *(const v8bf*)&Qs[row * 72 + kb + 16];
    }
    for (int jt = 0; jt < 4; ++jt)
      for (int ks = 0; ks < 2; ++ks) {
        Frag16 fb;
        int nrow = jt * 16 + l16, kb2 = hi * 8 + ks * 32;
        fb.h[0] = *(const v8bf*)&Ks[nrow * 72 + kb2];
        fb.h[1] = *(const v8bf*)&Ks[nrow * 72 + kb2 + 16];
        accS[jt] = wmma_bf16(fa[ks], fb, accS[jt]);
      }
    // dump S to LDS (row = mo + 8*hi + vgpr, col = jt*16 + l16)
    for (int jt = 0; jt < 4; ++jt)
      for (int i = 0; i < 8; ++i)
        Ss[(mo + hi * 8 + i) * 66 + jt * 16 + l16] = accS[jt][i];
    __syncthreads();

    // ---- online softmax, one thread per query row ----
    if (tid < 64) {
      const float scale = 0.125f;           // 1/sqrt(64)
      int m = tid;
      float tmax = -3.0e38f;
      for (int j = 0; j < 64; ++j) {
        float s = Ss[m * 66 + j] * scale + 1e-6f;
        if (mask_s[j] == 0) s = -1e9f;
        Ss[m * 66 + j] = s;
        tmax = fmaxf(tmax, s);
      }
      float mnew = fmaxf(m_run, tmax);
      float fac = __expf(m_run - mnew);
      float rsum = 0.f;
      for (int j = 0; j < 64; ++j) {
        float pv = __expf(Ss[m * 66 + j] - mnew);
        rsum += pv;
        Ps[m * 72 + j] = f2bf(pv);
      }
      s_run = s_run * fac + rsum;
      m_run = mnew;
      factor_s[m] = fac;
    }
    __syncthreads();

    // rescale O accumulators by per-row factor
    for (int dt = 0; dt < 4; ++dt)
      for (int i = 0; i < 8; ++i)
        accO[dt][i] *= factor_s[mo + hi * 8 + i];

    // ---- O += P V ----
    Frag16 pa[2];
    for (int ks = 0; ks < 2; ++ks) {
      int row = mo + l16, kb = hi * 8 + ks * 32;
      pa[ks].h[0] = *(const v8bf*)&Ps[row * 72 + kb];
      pa[ks].h[1] = *(const v8bf*)&Ps[row * 72 + kb + 16];
    }
    for (int dt = 0; dt < 4; ++dt)
      for (int ks = 0; ks < 2; ++ks) {
        Frag16 fb;
        int nrow = dt * 16 + l16, kb2 = hi * 8 + ks * 32;
        fb.h[0] = *(const v8bf*)&VTs[nrow * 72 + kb2];
        fb.h[1] = *(const v8bf*)&VTs[nrow * 72 + kb2 + 16];
        accO[dt] = wmma_bf16(pa[ks], fb, accO[dt]);
      }
  }

  if (tid < 64) rsum_s[tid] = 1.0f / s_run;
  __syncthreads();

  // store psi (token-major bf16 [b*Mq, 768]); optional V residual
  for (int dt = 0; dt < 4; ++dt) {
    int d = dt * 16 + l16;
    for (int i = 0; i < 8; ++i) {
      int mrow = mo + hi * 8 + i;
      int gm = m0 + mrow;
      if (gm >= Mq) continue;
      float o = accO[dt][i] * rsum_s[mrow];
      if (Vres) o += bf2f(Vres[((size_t)bh * Mq + gm) * DHEAD + d]);
      Out[(size_t)(b * Mq + gm) * DMODEL + h * DHEAD + d] = f2bf(o);
    }
  }
}

// ---------------------------------------------------------------------------
// out = LayerNorm(X + Res) * g + b   (one block per 768-wide row)
// ---------------------------------------------------------------------------
__launch_bounds__(256)
__global__ void ln_res_kernel(const float* __restrict__ X,
                              const float* __restrict__ Res,
                              const float* __restrict__ g,
                              const float* __restrict__ bta,
                              float* __restrict__ out) {
  __shared__ float red[256];
  const int row = blockIdx.x, tid = threadIdx.x;
  const float* xr = X + (size_t)row * DMODEL;
  const float* rr = Res + (size_t)row * DMODEL;
  float v0 = xr[tid]       + rr[tid];
  float v1 = xr[tid + 256] + rr[tid + 256];
  float v2 = xr[tid + 512] + rr[tid + 512];
  red[tid] = v0 + v1 + v2;
  __syncthreads();
  for (int off = 128; off > 0; off >>= 1) {
    if (tid < off) red[tid] += red[tid + off];
    __syncthreads();
  }
  float mean = red[0] * (1.0f / DMODEL);
  __syncthreads();
  float d0 = v0 - mean, d1 = v1 - mean, d2 = v2 - mean;
  red[tid] = d0 * d0 + d1 * d1 + d2 * d2;
  __syncthreads();
  for (int off = 128; off > 0; off >>= 1) {
    if (tid < off) red[tid] += red[tid + off];
    __syncthreads();
  }
  float rstd = rsqrtf(red[0] * (1.0f / DMODEL) + 1e-6f);
  float* orow = out + (size_t)row * DMODEL;
  orow[tid]       = d0 * rstd * g[tid]       + bta[tid];
  orow[tid + 256] = d1 * rstd * g[tid + 256] + bta[tid + 256];
  orow[tid + 512] = d2 * rstd * g[tid + 512] + bta[tid + 512];
}

// ---------------------------------------------------------------------------
// host launcher
// ---------------------------------------------------------------------------
extern "C" void kernel_launch(void* const* d_in, const int* in_sizes, int n_in,
                              void* d_out, int out_size, void* d_ws, size_t ws_size,
                              hipStream_t stream) {
  const int B = 4, L = 1024, R = 97;
  const int docTok = B * L;        // 4096
  const int relTok = B * R;        // 388
  const int BH = B * HNUM;         // 48

  const float* doc  = (const float*)d_in[0];
  const float* rel  = (const float*)d_in[1];
  const int*   mask = (const int*)d_in[2];
  const float* wq   = (const float*)d_in[3];
  const float* wk   = (const float*)d_in[4];
  const float* wv   = (const float*)d_in[5];
  const float* wkr  = (const float*)d_in[6];
  const float* wvr  = (const float*)d_in[7];
  const float* fcd  = (const float*)d_in[8];
  const float* fcr  = (const float*)d_in[9];
  const float* lng  = (const float*)d_in[10];
  const float* lnb  = (const float*)d_in[11];
  const float* lrg  = (const float*)d_in[12];
  const float* lrb  = (const float*)d_in[13];
  float* out = (float*)d_out;

  char* p = (char*)d_ws;
  auto alloc = [&](size_t bytes) -> char* {
    char* r = p;
    p += (bytes + 255) & ~(size_t)255;
    return r;
  };
  unsigned short* doc_bf = (unsigned short*)alloc((size_t)docTok * DMODEL * 2);
  unsigned short* rel_bf = (unsigned short*)alloc((size_t)relTok * DMODEL * 2);
  unsigned short* wqT  = (unsigned short*)alloc((size_t)DMODEL * DMODEL * 2);
  unsigned short* wkT  = (unsigned short*)alloc((size_t)DMODEL * DMODEL * 2);
  unsigned short* wvT  = (unsigned short*)alloc((size_t)DMODEL * DMODEL * 2);
  unsigned short* wkrT = (unsigned short*)alloc((size_t)DMODEL * DMODEL * 2);
  unsigned short* wvrT = (unsigned short*)alloc((size_t)DMODEL * DMODEL * 2);
  unsigned short* fcdT = (unsigned short*)alloc((size_t)DMODEL * DMODEL * 2);
  unsigned short* fcrT = (unsigned short*)alloc((size_t)DMODEL * DMODEL * 2);
  unsigned short* Qh = (unsigned short*)alloc((size_t)BH * L * DHEAD * 2);
  unsigned short* Kh = (unsigned short*)alloc((size_t)BH * L * DHEAD * 2);
  unsigned short* Vh = (unsigned short*)alloc((size_t)BH * L * DHEAD * 2);
  unsigned short* RKh = (unsigned short*)alloc((size_t)BH * R * DHEAD * 2);
  unsigned short* RVh = (unsigned short*)alloc((size_t)BH * R * DHEAD * 2);
  float* rksum = (float*)alloc((size_t)BH * DHEAD * 4);
  float* rvsum = (float*)alloc((size_t)BH * DHEAD * 4);
  unsigned short* FK = (unsigned short*)alloc((size_t)BH * L * DHEAD * 2);
  unsigned short* FV = (unsigned short*)alloc((size_t)BH * L * DHEAD * 2);
  unsigned short* psi_doc = (unsigned short*)alloc((size_t)docTok * DMODEL * 2);
  unsigned short* psi_rel = (unsigned short*)alloc((size_t)relTok * DMODEL * 2);
  float* fco_doc = (float*)alloc((size_t)docTok * DMODEL * 4);
  float* fco_rel = (float*)alloc((size_t)relTok * DMODEL * 4);

  // 1) convert activations / weights to bf16 (weights transposed N-major)
  {
    int nd = docTok * DMODEL, nr = relTok * DMODEL;
    cvt_bf16_kernel<<<(nd + 255) / 256, 256, 0, stream>>>(doc, doc_bf, nd);
    cvt_bf16_kernel<<<(nr + 255) / 256, 256, 0, stream>>>(rel, rel_bf, nr);
    int nw = DMODEL * DMODEL;
    dim3 gw((nw + 255) / 256);
    cvt_T_bf16_kernel<<<gw, 256, 0, stream>>>(wq, wqT);
    cvt_T_bf16_kernel<<<gw, 256, 0, stream>>>(wk, wkT);
    cvt_T_bf16_kernel<<<gw, 256, 0, stream>>>(wv, wvT);
    cvt_T_bf16_kernel<<<gw, 256, 0, stream>>>(wkr, wkrT);
    cvt_T_bf16_kernel<<<gw, 256, 0, stream>>>(wvr, wvrT);
    cvt_T_bf16_kernel<<<gw, 256, 0, stream>>>(fcd, fcdT);
    cvt_T_bf16_kernel<<<gw, 256, 0, stream>>>(fcr, fcrT);
  }

  // 2) projections (WMMA GEMMs) -> head-major bf16
  {
    dim3 gd((docTok + 127) / 128, DMODEL / 128);
    gemm_bf16_kernel<<<gd, 256, 0, stream>>>(doc_bf, wqT, Qh, docTok, 1, L);
    gemm_bf16_kernel<<<gd, 256, 0, stream>>>(doc_bf, wkT, Kh, docTok, 1, L);
    gemm_bf16_kernel<<<gd, 256, 0, stream>>>(doc_bf, wvT, Vh, docTok, 1, L);
    dim3 gr((relTok + 127) / 128, DMODEL / 128);
    gemm_bf16_kernel<<<gr, 256, 0, stream>>>(rel_bf, wkrT, RKh, relTok, 1, R);
    gemm_bf16_kernel<<<gr, 256, 0, stream>>>(rel_bf, wvrT, RVh, relTok, 1, R);
  }

  // 3) relation sums + fused K/V
  rel_sums_kernel<<<BH, 64, 0, stream>>>(RKh, RVh, rksum, rvsum, R);
  {
    int total = BH * L * DHEAD;
    fuse_kv_kernel<<<(total + 255) / 256, 256, 0, stream>>>(Kh, Vh, rksum, rvsum,
                                                            FK, FV, total);
  }

  // 4) attention (flash-style WMMA)
  {
    dim3 gdd(L / 64, BH);
    flash_attn_kernel<<<gdd, 128, 0, stream>>>(Qh, FK, FV, mask, nullptr,
                                               psi_doc, L);
    dim3 grr((R + 63) / 64, BH);
    flash_attn_kernel<<<grr, 128, 0, stream>>>(RKh, Kh, Vh, mask, RVh,
                                               psi_rel, R);
  }

  // 5) output FC GEMMs -> f32
  {
    dim3 gd((docTok + 127) / 128, DMODEL / 128);
    gemm_bf16_kernel<<<gd, 256, 0, stream>>>(psi_doc, fcdT, fco_doc, docTok, 2, L);
    dim3 gr((relTok + 127) / 128, DMODEL / 128);
    gemm_bf16_kernel<<<gr, 256, 0, stream>>>(psi_rel, fcrT, fco_rel, relTok, 2, R);
  }

  // 6) residual + LayerNorm into d_out (doc first, then rel)
  ln_res_kernel<<<docTok, 256, 0, stream>>>(fco_doc, doc, lng, lnb, out);
  ln_res_kernel<<<relTok, 256, 0, stream>>>(fco_rel, rel, lrg, lrb,
                                            out + (size_t)docTok * DMODEL);
}
